// VariationalDense_4252017623323
// MI455X (gfx1250) — matrix operand order repaired and checked by reading, not compiled, not run on hardware
//
#include <hip/hip_runtime.h>

// out = (x * bern) @ M + m  -- 4096^3 fp32 GEMM via bf16x3 (hi/lo split) WMMA.
// Pass 1: split x*bern and M^T into bf16 hi/lo planes in d_ws (one-time).
// Pass 2: WMMA GEMM; LDS double-buffered, staged with async global->LDS copies.

typedef __attribute__((ext_vector_type(16))) __bf16 v16bf;
typedef __attribute__((ext_vector_type(8)))  __bf16 bf8;
typedef __attribute__((ext_vector_type(8)))  float  v8f;
typedef __attribute__((ext_vector_type(4)))  unsigned short us4;

#define N_DIM 4096
#define BM 128
#define BN 128
#define BK 32
#define KT_CNT (N_DIM / BK)
#define LDT 40   // padded LDS row stride in halfs (80B = 20 banks -> conflict-free frag reads)

#if defined(__has_builtin)
#if __has_builtin(__builtin_amdgcn_global_load_async_to_lds_b128)
#define HAVE_ASYNC_B128 1
#endif
#endif

typedef int vint4 __attribute__((vector_size(16)));           // matches builtin pointee type
typedef __attribute__((address_space(1))) vint4 g_vint4;       // global AS source
typedef __attribute__((address_space(3))) vint4 l_vint4;       // LDS AS destination
typedef __attribute__((address_space(3))) void  lvoid;

static __device__ __forceinline__ unsigned short f32_to_bf16_rn(float f) {
  union { float f; unsigned u; } v; v.f = f;
  unsigned r = v.u + 0x7fffu + ((v.u >> 16) & 1u);   // round-to-nearest-even
  return (unsigned short)(r >> 16);
}
static __device__ __forceinline__ float bf16_to_f32(unsigned short s) {
  union { unsigned u; float f; } v; v.u = ((unsigned)s) << 16;
  return v.f;
}

static __device__ __forceinline__ void async_copy16(const unsigned short* gsrc,
                                                    unsigned short* ldst) {
#if defined(HAVE_ASYNC_B128)
  __builtin_amdgcn_global_load_async_to_lds_b128((g_vint4*)gsrc, (l_vint4*)ldst, 0, 0);
#else
  const unsigned lds_addr = (unsigned)(size_t)(lvoid*)ldst;
  asm volatile("global_load_async_to_lds_b128 %0, %1, off"
               :: "v"(lds_addr), "v"(gsrc) : "memory");
#endif
}
static __device__ __forceinline__ void wait_async0() {
#if __has_builtin(__builtin_amdgcn_s_wait_asynccnt)
  __builtin_amdgcn_s_wait_asynccnt(0);
#else
  asm volatile("s_wait_asynccnt 0" ::: "memory");
#endif
}

// ---------------- Pass 1a: A planes = split(x * bern), row-major ----------------
__global__ __launch_bounds__(256) void prep_x_split(
    const float* __restrict__ x, const float* __restrict__ bern,
    unsigned short* __restrict__ Ahi, unsigned short* __restrict__ Alo)
{
  const size_t i = ((size_t)blockIdx.x * 256 + threadIdx.x) * 4;
  const float4 xv = *(const float4*)(x + i);
  const float4 bv = *(const float4*)(bern + (i & (size_t)(N_DIM - 1)));
  const float f[4] = { xv.x * bv.x, xv.y * bv.y, xv.z * bv.z, xv.w * bv.w };
  us4 hv, lv;
#pragma unroll
  for (int j = 0; j < 4; ++j) {
    const unsigned short hs = f32_to_bf16_rn(f[j]);
    hv[j] = hs;
    lv[j] = f32_to_bf16_rn(f[j] - bf16_to_f32(hs));
  }
  *(us4*)(Ahi + i) = hv;
  *(us4*)(Alo + i) = lv;
}

// ------------- Pass 1b: B planes = split(M^T), tiled LDS transpose --------------
__global__ __launch_bounds__(256) void prep_mt_split(
    const float* __restrict__ Mw,
    unsigned short* __restrict__ Bhi, unsigned short* __restrict__ Blo)
{
  __shared__ float tile[64][68];
  const int tid = threadIdx.x;
  const int lx = tid & 15;   // float4 column
  const int ly = tid >> 4;   // 0..15
  const int kBase = blockIdx.y * 64;
  const int nBase = blockIdx.x * 64;
#pragma unroll
  for (int j = 0; j < 4; ++j) {
    const int r = ly + j * 16;
    const float4 v = *(const float4*)(Mw + (size_t)(kBase + r) * N_DIM + nBase + lx * 4);
    tile[r][lx * 4 + 0] = v.x; tile[r][lx * 4 + 1] = v.y;
    tile[r][lx * 4 + 2] = v.z; tile[r][lx * 4 + 3] = v.w;
  }
  __syncthreads();
#pragma unroll
  for (int j = 0; j < 4; ++j) {
    const int n = ly + j * 16;
    const int k = lx * 4;
    us4 hv, lv;
#pragma unroll
    for (int jj = 0; jj < 4; ++jj) {
      const float f = tile[k + jj][n];
      const unsigned short hs = f32_to_bf16_rn(f);
      hv[jj] = hs;
      lv[jj] = f32_to_bf16_rn(f - bf16_to_f32(hs));
    }
    *(us4*)(Bhi + (size_t)(nBase + n) * N_DIM + kBase + k) = hv;
    *(us4*)(Blo + (size_t)(nBase + n) * N_DIM + kBase + k) = lv;
  }
}

// --------- Pass 2: WMMA GEMM, async-staged LDS double buffer, bf16x3 -----------
__global__ __launch_bounds__(256) void vdense_gemm_staged(
    const unsigned short* __restrict__ Ahi, const unsigned short* __restrict__ Alo,
    const unsigned short* __restrict__ Bhi, const unsigned short* __restrict__ Blo,
    const float* __restrict__ mb, float* __restrict__ out)
{
  __shared__ unsigned short sm[2][4][BM * LDT];   // [buf][Ahi,Alo,Bhi,Blo][row*LDT+k]

  const int tid  = threadIdx.x;
  const int lane = tid & 31;
  const int wave = tid >> 5;
  const int lrow = lane & 15;
  const int hsel = lane >> 4;

  const int rowBase = blockIdx.y * BM;
  const int colBase = blockIdx.x * BN;
  const int wrow0 = (wave & 3) * 32;   // 4 wave-rows x 32 rows
  const int wcol0 = (wave >> 2) * 64;  // 2 wave-cols x 64 cols

  const unsigned short* gsrc[4]  = { Ahi, Alo, Bhi, Blo };
  const int             gbase[4] = { rowBase, rowBase, colBase, colBase };

  v8f acc[2][4];
#pragma unroll
  for (int mt = 0; mt < 2; ++mt)
#pragma unroll
    for (int nt = 0; nt < 4; ++nt)
      acc[mt][nt] = v8f{};

  // 8 x 16B async copies per thread per K tile (4 planes x 128 rows x 64B)
  auto stage = [&](int kt, int b) {
    const int k0 = kt * BK;
#pragma unroll
    for (int p = 0; p < 4; ++p)
#pragma unroll
      for (int t = 0; t < 2; ++t) {
        const int pc  = tid + t * 256;   // 0..511 chunk id within plane
        const int row = pc >> 2;         // 0..127
        const int c   = (pc & 3) * 8;    // half offset within row: 0,8,16,24
        const unsigned short* g = gsrc[p] + (size_t)(gbase[p] + row) * N_DIM + (k0 + c);
        async_copy16(g, &sm[b][p][row * LDT + c]);
      }
  };

  stage(0, 0);

  for (int kt = 0; kt < KT_CNT; ++kt) {
    wait_async0();        // this wave's copies for buf[kt&1] are in LDS
    __syncthreads();      // all waves' copies visible; previous compute finished
    if (kt + 1 < KT_CNT)
      stage(kt + 1, (kt + 1) & 1);   // prefetch next tile while computing

    const int b = kt & 1;
    const unsigned short* sAhi = sm[b][0];
    const unsigned short* sAlo = sm[b][1];
    const unsigned short* sBhi = sm[b][2];
    const unsigned short* sBlo = sm[b][3];

    union Frag { v16bf v; bf8 h[2]; };
    Frag fAhi[2], fAlo[2], fBhi[4], fBlo[4];
#pragma unroll
    for (int mt = 0; mt < 2; ++mt) {
      // 16-bit A 16x32 layout: lane holds row lane%16; K chunks [b..b+7],[b+16..b+23], b=8*hsel
      const int ar = (wrow0 + mt * 16 + lrow) * LDT + hsel * 8;
      fAhi[mt].h[0] = *(const bf8*)&sAhi[ar];
      fAhi[mt].h[1] = *(const bf8*)&sAhi[ar + 16];
      fAlo[mt].h[0] = *(const bf8*)&sAlo[ar];
      fAlo[mt].h[1] = *(const bf8*)&sAlo[ar + 16];
    }
#pragma unroll
    for (int nt = 0; nt < 4; ++nt) {
      // B 32x16: lane holds column lane%16; K chunks [b..b+7],[b+16..b+23], b=8*hsel
      const int br = (wcol0 + nt * 16 + lrow) * LDT + hsel * 8;
      fBhi[nt].h[0] = *(const bf8*)&sBhi[br];
      fBhi[nt].h[1] = *(const bf8*)&sBhi[br + 16];
      fBlo[nt].h[0] = *(const bf8*)&sBlo[br];
      fBlo[nt].h[1] = *(const bf8*)&sBlo[br + 16];
    }

#pragma unroll
    for (int mt = 0; mt < 2; ++mt)
#pragma unroll
      for (int nt = 0; nt < 4; ++nt) {
        acc[mt][nt] = __builtin_amdgcn_wmma_f32_16x16x32_bf16(
            false, fAhi[mt].v, false, fBhi[nt].v, (short)0, acc[mt][nt], false, false);
        acc[mt][nt] = __builtin_amdgcn_wmma_f32_16x16x32_bf16(
            false, fAlo[mt].v, false, fBhi[nt].v, (short)0, acc[mt][nt], false, false);
        acc[mt][nt] = __builtin_amdgcn_wmma_f32_16x16x32_bf16(
            false, fAhi[mt].v, false, fBlo[nt].v, (short)0, acc[mt][nt], false, false);
      }
  }

  // epilogue: + m[col]; C/D layout: VGPR v -> row v + 8*hsel, lane%16 -> col
  const int orow0 = rowBase + wrow0;
  const int ocol0 = colBase + wcol0;
  float bias[4];
#pragma unroll
  for (int nt = 0; nt < 4; ++nt) bias[nt] = mb[ocol0 + nt * 16 + lrow];
#pragma unroll
  for (int mt = 0; mt < 2; ++mt)
#pragma unroll
    for (int nt = 0; nt < 4; ++nt)
#pragma unroll
      for (int v = 0; v < 8; ++v) {
        const int r = orow0 + mt * 16 + v + 8 * hsel;
        out[(size_t)r * N_DIM + (ocol0 + nt * 16 + lrow)] = acc[mt][nt][v] + bias[nt];
      }
}

// ---------------- Fallback: fused-conversion kernel (small workspace) ----------
__global__ __launch_bounds__(256) void vdense_gemm_bf16x3(
    const float* __restrict__ x, const float* __restrict__ bern,
    const float* __restrict__ Mw, const float* __restrict__ mb,
    float* __restrict__ out)
{
  __shared__ unsigned short sAhi[BM * LDT];
  __shared__ unsigned short sAlo[BM * LDT];
  __shared__ unsigned short sBhi[BN * LDT];
  __shared__ unsigned short sBlo[BN * LDT];

  const int tid  = threadIdx.x;
  const int lane = tid & 31;
  const int wave = tid >> 5;
  const int lrow = lane & 15;
  const int hsel = lane >> 4;
  const int rowBase = blockIdx.y * BM;
  const int colBase = blockIdx.x * BN;
  const int wrow0 = (wave & 3) * 32;
  const int wcol0 = (wave >> 2) * 64;

  v8f acc[2][4];
#pragma unroll
  for (int mt = 0; mt < 2; ++mt)
#pragma unroll
    for (int nt = 0; nt < 4; ++nt)
      acc[mt][nt] = v8f{};

  float4 ax[4], abn[4];
  float  bmv[4][4];

  for (int kt = 0; kt < KT_CNT; ++kt) {
    const int k0 = kt * BK;
#pragma unroll
    for (int t = 0; t < 4; ++t) {
      const int chunk = tid + t * 256;
      const int row   = chunk >> 3;
      const int k     = (chunk & 7) << 2;
      ax[t]  = *(const float4*)(x + (size_t)(rowBase + row) * N_DIM + (k0 + k));
      abn[t] = *(const float4*)(bern + (k0 + k));
    }
#pragma unroll
    for (int t = 0; t < 4; ++t) {
      const int chunk = tid + t * 256;
      const int n     = chunk & 127;
      const int kc    = chunk >> 7;
#pragma unroll
      for (int j = 0; j < 4; ++j)
        bmv[t][j] = Mw[(size_t)(k0 + kc * 4 + j) * N_DIM + (colBase + n)];
    }
    __syncthreads();
#pragma unroll
    for (int t = 0; t < 4; ++t) {
      const int chunk = tid + t * 256;
      const int row   = chunk >> 3;
      const int k     = (chunk & 7) << 2;
      const float f[4] = { ax[t].x * abn[t].x, ax[t].y * abn[t].y,
                           ax[t].z * abn[t].z, ax[t].w * abn[t].w };
      us4 hv, lv;
#pragma unroll
      for (int j = 0; j < 4; ++j) {
        const unsigned short hs = f32_to_bf16_rn(f[j]);
        hv[j] = hs;
        lv[j] = f32_to_bf16_rn(f[j] - bf16_to_f32(hs));
      }
      *(us4*)&sAhi[row * LDT + k] = hv;
      *(us4*)&sAlo[row * LDT + k] = lv;
    }
#pragma unroll
    for (int t = 0; t < 4; ++t) {
      const int chunk = tid + t * 256;
      const int n     = chunk & 127;
      const int k     = (chunk >> 7) << 2;
      us4 hv, lv;
#pragma unroll
      for (int j = 0; j < 4; ++j) {
        const unsigned short hs = f32_to_bf16_rn(bmv[t][j]);
        hv[j] = hs;
        lv[j] = f32_to_bf16_rn(bmv[t][j] - bf16_to_f32(hs));
      }
      *(us4*)&sBhi[n * LDT + k] = hv;
      *(us4*)&sBlo[n * LDT + k] = lv;
    }
    __syncthreads();

    union Frag { v16bf v; bf8 h[2]; };
    Frag fAhi[2], fAlo[2], fBhi[4], fBlo[4];
#pragma unroll
    for (int mt = 0; mt < 2; ++mt) {
      const int ar = (wrow0 + mt * 16 + lrow) * LDT + hsel * 8;
      fAhi[mt].h[0] = *(const bf8*)&sAhi[ar];
      fAhi[mt].h[1] = *(const bf8*)&sAhi[ar + 16];
      fAlo[mt].h[0] = *(const bf8*)&sAlo[ar];
      fAlo[mt].h[1] = *(const bf8*)&sAlo[ar + 16];
    }
#pragma unroll
    for (int nt = 0; nt < 4; ++nt) {
      const int br = (wcol0 + nt * 16 + lrow) * LDT + hsel * 8;
      fBhi[nt].h[0] = *(const bf8*)&sBhi[br];
      fBhi[nt].h[1] = *(const bf8*)&sBhi[br + 16];
      fBlo[nt].h[0] = *(const bf8*)&sBlo[br];
      fBlo[nt].h[1] = *(const bf8*)&sBlo[br + 16];
    }
#pragma unroll
    for (int mt = 0; mt < 2; ++mt)
#pragma unroll
      for (int nt = 0; nt < 4; ++nt) {
        acc[mt][nt] = __builtin_amdgcn_wmma_f32_16x16x32_bf16(
            false, fAhi[mt].v, false, fBhi[nt].v, (short)0, acc[mt][nt], false, false);
        acc[mt][nt] = __builtin_amdgcn_wmma_f32_16x16x32_bf16(
            false, fAlo[mt].v, false, fBhi[nt].v, (short)0, acc[mt][nt], false, false);
        acc[mt][nt] = __builtin_amdgcn_wmma_f32_16x16x32_bf16(
            false, fAhi[mt].v, false, fBlo[nt].v, (short)0, acc[mt][nt], false, false);
      }
  }

  const int orow0 = rowBase + wrow0;
  const int ocol0 = colBase + wcol0;
  float bias[4];
#pragma unroll
  for (int nt = 0; nt < 4; ++nt) bias[nt] = mb[ocol0 + nt * 16 + lrow];
#pragma unroll
  for (int mt = 0; mt < 2; ++mt)
#pragma unroll
    for (int nt = 0; nt < 4; ++nt)
#pragma unroll
      for (int v = 0; v < 8; ++v) {
        const int r = orow0 + mt * 16 + v + 8 * hsel;
        out[(size_t)r * N_DIM + (ocol0 + nt * 16 + lrow)] = acc[mt][nt][v] + bias[nt];
      }
}

extern "C" void kernel_launch(void* const* d_in, const int* in_sizes, int n_in,
                              void* d_out, int out_size, void* d_ws, size_t ws_size,
                              hipStream_t stream) {
  (void)in_sizes; (void)n_in; (void)out_size;
  const float* x    = (const float*)d_in[0];
  const float* bern = (const float*)d_in[1];
  const float* Mw   = (const float*)d_in[2];
  const float* mb   = (const float*)d_in[3];
  float* out = (float*)d_out;

  const size_t PLANE = (size_t)N_DIM * N_DIM;       // elements per plane
  const size_t NEED  = PLANE * sizeof(unsigned short) * 4;

  if (ws_size >= NEED) {
    unsigned short* Ahi = (unsigned short*)d_ws;
    unsigned short* Alo = Ahi + PLANE;
    unsigned short* Bhi = Alo + PLANE;
    unsigned short* Blo = Bhi + PLANE;
    prep_x_split<<<(unsigned)(PLANE / 4 / 256), 256, 0, stream>>>(x, bern, Ahi, Alo);
    prep_mt_split<<<dim3(N_DIM / 64, N_DIM / 64), 256, 0, stream>>>(Mw, Bhi, Blo);
    vdense_gemm_staged<<<dim3(N_DIM / BN, N_DIM / BM), 256, 0, stream>>>(
        Ahi, Alo, Bhi, Blo, mb, out);
  } else {
    vdense_gemm_bf16x3<<<dim3(N_DIM / BN, N_DIM / BM), 256, 0, stream>>>(
        x, bern, Mw, mb, out);
  }
}
